// VocabGraphConvolution_12876311953623
// MI455X (gfx1250) — matrix operand order repaired and checked by reading, not compiled, not run on hardware
//
#include <hip/hip_runtime.h>
#include <hip/hip_bf16.h>

// Problem constants (match reference)
#define VOC   30000
#define HID   256
#define OUTD  256
#define D_DIM 1024
#define NNZ_  1500000

typedef __attribute__((ext_vector_type(16))) _Float16 v16h;
typedef __attribute__((ext_vector_type(8)))  _Float16 v8h;
typedef __attribute__((ext_vector_type(4)))  _Float16 v4h;
typedef __attribute__((ext_vector_type(8)))  float    v8f;
typedef __attribute__((ext_vector_type(4)))  float    v4f;

// ---------------------------------------------------------------------------
// Kernel 0: zero the workspace accumulators (Hsum + fused) every call.
// ---------------------------------------------------------------------------
__global__ __launch_bounds__(256) void k_zero(float* __restrict__ p, int n4) {
    int i = blockIdx.x * blockDim.x + threadIdx.x;
    if (i < n4) {
        v4f z = {0.f, 0.f, 0.f, 0.f};
        *(v4f*)(p + (size_t)i * 4) = z;
    }
}

// ---------------------------------------------------------------------------
// Kernel 1: sparse scatter  Hsum[row,:] += val * W[col,:]
// One wave (32 lanes) per nnz; 8 coalesced f32 atomics per lane.
// Hsum (30 MB) is L2-resident on MI455X (192 MB L2).
// ---------------------------------------------------------------------------
__global__ __launch_bounds__(256) void k_scatter(
    const int*   __restrict__ rows,
    const int*   __restrict__ cols,
    const float* __restrict__ vals,
    const float* __restrict__ W,
    float*       __restrict__ Hsum,
    int nnz)
{
    int wave = blockIdx.x * (blockDim.x >> 5) + (threadIdx.x >> 5);
    int lane = threadIdx.x & 31;
    if (wave >= nnz) return;
    int   r = rows[wave];
    int   c = cols[wave];
    float v = vals[wave];
    const float* wrow = W    + (size_t)c * HID;
    float*       hrow = Hsum + (size_t)r * HID;
#pragma unroll
    for (int j = 0; j < 8; ++j) {
        int col = lane + 32 * j;                 // coalesced across the wave
        atomicAdd(&hrow[col], v * wrow[col]);
    }
}

// ---------------------------------------------------------------------------
// Kernel 2: fused += X[1024 x 30000] @ Hsum[30000 x 256]   (f16 WMMA, f32 acc)
// 128 threads = 4 waves, 64(M) x 64(N) tile, K chunked by 32, K-split 16
// with f32 atomic reduction. Software-pipelined, double-buffered LDS:
//   store_lds(p) -> barrier -> global loads (chunk+1) -> wmma(p)
// A in LDS row-major [m][k]; B transposed [n][k]; all LDS traffic is b128.
// Only the globally-last K-chunk (K tail = 16) needs bounds guards; it is
// peeled out of the pipelined loop.
// ---------------------------------------------------------------------------
#define BK        32
#define LDP       40          // row pitch in f16 (80 B, 16B-aligned)
#define NCHUNK    ((VOC + BK - 1) / BK)   // 938 (chunk 937 is half-full)
#define KSPLIT    16
#define CH_PER_SP ((NCHUNK + KSPLIT - 1) / KSPLIT)  // 59

struct StageRegs {
    v4f   a[2][2];   // A: 2 tasks x 8 consecutive K floats
    float b[16];     // B: 16-deep K segment of one N column
};

template<bool GUARD>
__device__ __forceinline__ void load_regs(
    const float* __restrict__ X,
    const float* __restrict__ Hsum,
    int kc, int m0, int n0, int t, StageRegs& rg)
{
    const v4f z = {0.f, 0.f, 0.f, 0.f};
    // ---- A: task fid covers row m = fid>>2, K octet ko = (fid&3)*8.
    //      4 consecutive threads read 32 consecutive floats (128B, coalesced).
#pragma unroll
    for (int i = 0; i < 2; ++i) {
        const int fid = t + 128 * i;
        const int m   = fid >> 2;
        const int ko  = (fid & 3) * 8;
        const float* src = X + (size_t)(m0 + m) * VOC + kc + ko;
#pragma unroll
        for (int h = 0; h < 2; ++h) {
            if (!GUARD || (kc + ko + 4 * h < VOC))    // VOC % 4 == 0
                rg.a[i][h] = *(const v4f*)(src + 4 * h);
            else
                rg.a[i][h] = z;
        }
        if (!GUARD)
            __builtin_prefetch(src + 2 * BK, 0, 3);   // global_prefetch_b8
    }
    // ---- B: n = t&63, kq = (t>>6)*16; each of the 16 loads is a fully
    //      coalesced 128B wave request (consecutive n across lanes).
    const int n  = t & 63;
    const int kq = (t >> 6) * 16;
    const float* bsrc = Hsum + (size_t)(kc + kq) * HID + n0 + n;
#pragma unroll
    for (int j = 0; j < 16; ++j) {
        if (!GUARD || (kc + kq + j < VOC))
            rg.b[j] = bsrc[(size_t)j * HID];
        else
            rg.b[j] = 0.f;
    }
    if (!GUARD)
        __builtin_prefetch(bsrc + (size_t)2 * BK * HID, 0, 3);
}

__device__ __forceinline__ void store_lds(
    const StageRegs& rg, int t,
    _Float16 (*As)[LDP],     // [64][LDP], [m][k]
    _Float16 (*Bs)[LDP])     // [64][LDP], [n][k] (transposed)
{
#pragma unroll
    for (int i = 0; i < 2; ++i) {
        const int fid = t + 128 * i;
        const int m   = fid >> 2;
        const int ko  = (fid & 3) * 8;
        v8h h;
#pragma unroll
        for (int q = 0; q < 2; ++q)
#pragma unroll
            for (int j = 0; j < 4; ++j)
                h[q * 4 + j] = (_Float16)rg.a[i][q][j];
        *(v8h*)&As[m][ko] = h;                        // single ds_store_b128
    }
    const int n  = t & 63;
    const int kq = (t >> 6) * 16;
#pragma unroll
    for (int half = 0; half < 2; ++half) {
        v8h h;
#pragma unroll
        for (int j = 0; j < 8; ++j)
            h[j] = (_Float16)rg.b[half * 8 + j];
        *(v8h*)&Bs[n][kq + half * 8] = h;             // single ds_store_b128
    }
}

__device__ __forceinline__ void wmma_step(
    const _Float16 (*As)[LDP],
    const _Float16 (*Bs)[LDP],
    int w, int g, int l16, v8f (&acc)[4])
{
    // A fragment: lane l16 = row, VGPRs hold K = {g*8, 16+g*8} + [0..7]
    const int ml = w * 16 + l16;
    v8h alo = *(const v8h*)&As[ml][g * 8];
    v8h ahi = *(const v8h*)&As[ml][16 + g * 8];
    v16h a;
#pragma unroll
    for (int i = 0; i < 8; ++i) { a[i] = alo[i]; a[8 + i] = ahi[i]; }

#pragma unroll
    for (int s = 0; s < 4; ++s) {
        // B fragment: lane l16 = col, VGPRs hold K = {g*8, 16+g*8} + [0..7]
        const int nl = s * 16 + l16;
        v8h blo = *(const v8h*)&Bs[nl][g * 8];
        v8h bhi = *(const v8h*)&Bs[nl][16 + g * 8];
        v16h b;
#pragma unroll
        for (int i = 0; i < 8; ++i) { b[i] = blo[i]; b[8 + i] = bhi[i]; }

        acc[s] = __builtin_amdgcn_wmma_f32_16x16x32_f16(
            false, a, false, b, (short)0, acc[s], false, false);
    }
}

__global__ __launch_bounds__(128) void k_gemm_main(
    const float* __restrict__ X,     // [D_DIM, VOC]
    const float* __restrict__ Hsum,  // [VOC, HID]
    float*       __restrict__ fused) // [D_DIM, HID] (atomic accum)
{
    __shared__ __align__(16) _Float16 As[2][64][LDP];
    __shared__ __align__(16) _Float16 Bs[2][64][LDP];

    const int t    = threadIdx.x;
    const int lane = t & 31;
    const int w    = t >> 5;          // wave id 0..3 -> M substrip
    const int m0   = blockIdx.x * 64;
    const int n0   = blockIdx.y * 64;
    const int g    = lane >> 4;       // lane half
    const int l16  = lane & 15;

    const int ch_begin = blockIdx.z * CH_PER_SP;
    int ch_end = ch_begin + CH_PER_SP;
    if (ch_end > NCHUNK) ch_end = NCHUNK;
    const int  ch_full  = (ch_end < NCHUNK) ? ch_end : (NCHUNK - 1);
    const bool has_tail = (ch_end == NCHUNK);
    // note: every K-split slice contains >= 52 full chunks, so the pipelined
    // loop below (which assumes at least 1 full chunk + prologue) is safe.

    v8f acc[4];
#pragma unroll
    for (int s = 0; s < 4; ++s)
#pragma unroll
        for (int r = 0; r < 8; ++r) acc[s][r] = 0.f;

    StageRegs rg;
    int p = 0;

    // prologue: fetch first chunk into registers
    load_regs<false>(X, Hsum, ch_begin * BK, m0, n0, t, rg);

    // steady state: double-buffered, one barrier per iteration
    for (int ch = ch_begin; ch < ch_full - 1; ++ch) {
        store_lds(rg, t, As[p], Bs[p]);
        __syncthreads();
        load_regs<false>(X, Hsum, (ch + 1) * BK, m0, n0, t, rg);  // overlaps wmma
        wmma_step(As[p], Bs[p], w, g, l16, acc);
        p ^= 1;
    }
    // last full chunk (+ optional guarded fetch of the partial tail chunk)
    store_lds(rg, t, As[p], Bs[p]);
    __syncthreads();
    if (has_tail)
        load_regs<true>(X, Hsum, (NCHUNK - 1) * BK, m0, n0, t, rg);
    wmma_step(As[p], Bs[p], w, g, l16, acc);
    p ^= 1;
    if (has_tail) {
        store_lds(rg, t, As[p], Bs[p]);
        __syncthreads();
        wmma_step(As[p], Bs[p], w, g, l16, acc);
    }

    // ---- K-split reduction: atomic add the 64x64 f32 tile into fused.
#pragma unroll
    for (int s = 0; s < 4; ++s) {
        const int col = n0 + s * 16 + l16;
#pragma unroll
        for (int r = 0; r < 8; ++r) {
            const int row = m0 + w * 16 + g * 8 + r;
            atomicAdd(&fused[(size_t)row * HID + col], acc[s][r]);
        }
    }
}

// ---------------------------------------------------------------------------
// Kernel 3: out = fused[1024 x 256] @ fc_w[256 x 256] + fc_b   (f16 WMMA)
// One wave per 16x16 output tile; K = 256 = 8 chunks of 32.
// ---------------------------------------------------------------------------
__global__ __launch_bounds__(32) void k_gemm_fc(
    const float* __restrict__ fused,
    const float* __restrict__ fc_w,   // [HID, OUTD]
    const float* __restrict__ fc_b,
    float*       __restrict__ out)    // [D_DIM, OUTD]
{
    const int lane = threadIdx.x & 31;
    const int g    = lane >> 4;
    const int l16  = lane & 15;
    const int m0   = blockIdx.x * 16;
    const int n0   = blockIdx.y * 16;

    v8f acc;
#pragma unroll
    for (int r = 0; r < 8; ++r) acc[r] = 0.f;

    const int m = m0 + l16;
    const int n = n0 + l16;

#pragma unroll
    for (int kc = 0; kc < HID; kc += 32) {
        // A fragment from fused (row-major, K contiguous): two f32x4 pairs
        v16h a;
#pragma unroll
        for (int h = 0; h < 2; ++h) {
            const int kb = kc + 16 * h + g * 8;
            v4f f0 = *(const v4f*)(fused + (size_t)m * HID + kb);
            v4f f1 = *(const v4f*)(fused + (size_t)m * HID + kb + 4);
#pragma unroll
            for (int i = 0; i < 4; ++i) {
                a[8 * h + i]     = (_Float16)f0[i];
                a[8 * h + 4 + i] = (_Float16)f1[i];
            }
        }
        // B fragment from fc_w ([K][N] row-major): 16 strided scalars,
        // each coalesced across the 16 lanes of a half-wave.
        v16h b;
#pragma unroll
        for (int i = 0; i < 16; ++i)
            b[i] = (_Float16)fc_w[(size_t)(kc + g * 8 + (i & 7) + (i >> 3) * 16) * OUTD + n];

        acc = __builtin_amdgcn_wmma_f32_16x16x32_f16(
            false, a, false, b, (short)0, acc, false, false);
    }

    const float bias = fc_b[n];
#pragma unroll
    for (int r = 0; r < 8; ++r) {
        const int row = m0 + g * 8 + r;
        out[(size_t)row * OUTD + n] = acc[r] + bias;
    }
}

// ---------------------------------------------------------------------------
// Host-side launcher
// ---------------------------------------------------------------------------
extern "C" void kernel_launch(void* const* d_in, const int* in_sizes, int n_in,
                              void* d_out, int out_size, void* d_ws, size_t ws_size,
                              hipStream_t stream) {
    const int*   a0r  = (const int*)  d_in[0];
    const int*   a0c  = (const int*)  d_in[1];
    const float* a0v  = (const float*)d_in[2];
    const int*   a1r  = (const int*)  d_in[3];
    const int*   a1c  = (const int*)  d_in[4];
    const float* a1v  = (const float*)d_in[5];
    const float* X    = (const float*)d_in[6];
    const float* W0   = (const float*)d_in[7];
    const float* W1   = (const float*)d_in[8];
    const float* fcw  = (const float*)d_in[9];
    const float* fcb  = (const float*)d_in[10];
    float*       out  = (float*)d_out;

    float* Hsum  = (float*)d_ws;                       // [VOC, HID]
    float* fused = Hsum + (size_t)VOC * HID;           // [D_DIM, HID]

    // 0) zero accumulators (Hsum + fused) — deterministic per call
    const int nz4 = ((size_t)VOC * HID + (size_t)D_DIM * HID) / 4;
    k_zero<<<(nz4 + 255) / 256, 256, 0, stream>>>(Hsum, nz4);

    // 1) scatter both sparse branches into the shared accumulator
    const int nnz = in_sizes[0];
    const int sblocks = (nnz + 7) / 8;                 // 8 waves per 256-thr block
    k_scatter<<<sblocks, 256, 0, stream>>>(a0r, a0c, a0v, W0, Hsum, nnz);
    k_scatter<<<sblocks, 256, 0, stream>>>(a1r, a1c, a1v, W1, Hsum, nnz);

    // 2) fused = X @ Hsum  (WMMA, K-split 16, pipelined)
    dim3 g2(D_DIM / 64, HID / 64, KSPLIT);
    k_gemm_main<<<g2, 128, 0, stream>>>(X, Hsum, fused);

    // 3) out = fused @ fc_w + fc_b  (WMMA)
    dim3 g3(D_DIM / 16, OUTD / 16);
    k_gemm_fc<<<g3, 32, 0, stream>>>(fused, fcw, fcb, out);
}
